// BlockSparseAttention_19490561589551
// MI455X (gfx1250) — matrix operand order repaired and checked by reading, not compiled, not run on hardware
//
#include <hip/hip_runtime.h>

typedef __bf16 bf16_t;
typedef __attribute__((ext_vector_type(16))) __bf16 v16bf;
typedef __attribute__((ext_vector_type(8)))  __bf16 v8bf;
typedef __attribute__((ext_vector_type(8)))  float  v8f;

#define WMMA_BF16(A,B,C) __builtin_amdgcn_wmma_f32_16x16x32_bf16( \
    false, (A), false, (B), (short)0, (C), false, false)

// Problem constants (from reference): B=4, N=4096, C=1024, H=16, d=64, bs=64
constexpr int Bsz = 4;
constexpr int N_  = 4096;
constexpr int C_  = 1024;
constexpr int H_  = 16;
constexpr int D_  = 64;
constexpr int BS  = 64;
constexpr int NB  = 64;          // N / BS
constexpr int M_  = Bsz * N_;    // 16384 rows for all GEMMs
constexpr int K_  = C_;          // 1024

// ---------------------------------------------------------------- converts
__global__ void cvt_f32_bf16(const float* __restrict__ src,
                             bf16_t* __restrict__ dst, int n) {
  int i = blockIdx.x * blockDim.x + threadIdx.x;
  if (i < n) dst[i] = (bf16_t)src[i];
}

// ---------------------------------------------------------------- GEMM
// Y = X @ W^T + bias.  X: [M,K] bf16 row-major, W: [N=1024,K] bf16 row-major.
// Wave tile 32(M) x 64(N) = 2x4 WMMA tiles: 8 WMMAs per 6 operand-tile loads
// per 32-wide k-step (~21 flops/byte from L2).  8 waves = 128x128 per wg.
// mode 0: store bf16 into [B,H,N,d]          (Q, K)
// mode 1: store bf16 into [B,H,nb,d,bs]      (V transposed per block)
// mode 2: store f32  into [M, C] row-major   (final output)
__global__ __launch_bounds__(256) void gemm_wmma(
    const bf16_t* __restrict__ X, const bf16_t* __restrict__ W,
    const float* __restrict__ bias, void* __restrict__ outp, int mode)
{
  const int tid  = threadIdx.x;
  const int lane = tid & 31;
  const int wave = tid >> 5;          // 8 waves: 4 along M, 2 along N
  const int wm = wave >> 1, wn = wave & 1;
  const int m0 = blockIdx.x * 128 + wm * 32;
  const int n0 = blockIdx.y * 128 + wn * 64;

  const int cl = lane & 15;           // A row / B col within tile
  const int lh = lane >> 4;           // lane-half

  v8f acc[2][4] = {};

  const bf16_t* ar[2];
  ar[0] = X + (size_t)(m0 + cl) * K_;
  ar[1] = ar[0] + (size_t)16 * K_;
  const bf16_t* br[4];
  br[0] = W + (size_t)(n0 + cl) * K_;
  br[1] = br[0] + (size_t)16 * K_;
  br[2] = br[0] + (size_t)32 * K_;
  br[3] = br[0] + (size_t)48 * K_;
  const int ka = lh * 16;             // A chunk offset within k-step
  const int kb = lh * 16;             // B chunk offset within k-step

  for (int k = 0; k < K_; k += 32) {
    union { v16bf v; v8bf hl[2]; } a[2], b[4];
    #pragma unroll
    for (int i = 0; i < 2; ++i) {
      a[i].hl[0] = *(const v8bf*)(ar[i] + k + ka);
      a[i].hl[1] = *(const v8bf*)(ar[i] + k + ka + 8);
    }
    #pragma unroll
    for (int j = 0; j < 4; ++j) {
      b[j].hl[0] = *(const v8bf*)(br[j] + k + kb);
      b[j].hl[1] = *(const v8bf*)(br[j] + k + kb + 8);
    }
    #pragma unroll
    for (int i = 0; i < 2; ++i)
      #pragma unroll
      for (int j = 0; j < 4; ++j)
        acc[i][j] = WMMA_BF16(a[i].v, b[j].v, acc[i][j]);
  }

  #pragma unroll
  for (int im = 0; im < 2; ++im)
  #pragma unroll
  for (int in = 0; in < 4; ++in)
  #pragma unroll
  for (int r = 0; r < 8; ++r) {
    const int row = m0 + im * 16 + lh * 8 + r;   // C-tile row mapping
    const int col = n0 + in * 16 + cl;
    float v = acc[im][in][r] + bias[col];
    if (mode == 2) {
      ((float*)outp)[(size_t)row * C_ + col] = v;
    } else {
      const int b  = row >> 12, tok = row & (N_ - 1);
      const int hh = col >> 6,  dd  = col & 63;
      if (mode == 0) {
        ((bf16_t*)outp)[(((size_t)(b * H_ + hh) * N_ + tok) << 6) + dd] = (bf16_t)v;
      } else {
        const int blk = tok >> 6, trow = tok & 63;
        ((bf16_t*)outp)[((((size_t)(b * H_ + hh) * NB + blk) * D_ + dd) << 6) + trow] = (bf16_t)v;
      }
    }
  }
}

// ---------------------------------------------------------------- attention
// One workgroup (4 waves / 128 thr) per (b,h,block). Each wave owns 16 query
// rows. S = q @ kw^T (WMMA), register softmax with shfl_xor row reductions,
// P->bf16 via LDS (C-layout -> A-layout transpose), O = P @ vw (WMMA).
__global__ __launch_bounds__(128) void attn_wmma(
    const bf16_t* __restrict__ Q, const bf16_t* __restrict__ Km,
    const bf16_t* __restrict__ Vt, bf16_t* __restrict__ O)
{
  __shared__ __align__(16) bf16_t P[4][16][3 * BS];   // 24 KB

  const int tid  = threadIdx.x;
  const int lane = tid & 31;
  const int wq   = tid >> 5;          // wave index = query sub-tile
  const int lh   = lane >> 4;
  const int cl   = lane & 15;

  const int bid = blockIdx.x;         // B*H*NB = 4096
  const int blk = bid & (NB - 1);
  const int hh  = (bid >> 6) & (H_ - 1);
  const int b   = bid >> 10;

  const size_t headoff = ((size_t)(b * H_ + hh) * N_) << 6;   // *d
  const bf16_t* qh = Q  + headoff;
  const bf16_t* kh = Km + headoff;

  const int q0 = blk * BS + wq * 16;
  const bf16_t* qrow = qh + (size_t)(q0 + cl) * D_;

  const bool leftOK  = (blk > 0);
  const bool rightOK = (blk < NB - 1);

  // ---- S = q @ kw^T : 12 N-tiles of 16 keys, k-dim = d = 64 (2 steps of 32)
  v8f s[12];
  #pragma unroll
  for (int t = 0; t < 12; ++t) s[t] = v8f{};

  const int tlo = leftOK  ? 0 : 4;    // uniform across workgroup
  const int thi = rightOK ? 12 : 8;
  for (int ks = 0; ks < 2; ++ks) {
    union { v16bf v; v8bf hl[2]; } a;
    a.hl[0] = *(const v8bf*)(qrow + ks * 32 + lh * 16);
    a.hl[1] = *(const v8bf*)(qrow + ks * 32 + lh * 16 + 8);
    for (int t = tlo; t < thi; ++t) {
      const int jtok = (blk - 1) * BS + t * 16 + cl;   // key token index
      const bf16_t* krow = kh + (size_t)jtok * D_ + ks * 32 + lh * 16;
      union { v16bf v; v8bf hl[2]; } bm;
      bm.hl[0] = *(const v8bf*)(krow);
      bm.hl[1] = *(const v8bf*)(krow + 8);
      s[t] = WMMA_BF16(a.v, bm.v, s[t]);
    }
  }

  // ---- mask + softmax over the 192-wide window, rows in C-layout
  const float scale = 0.125f;                 // d^-0.5, d = 64
  const float NEG = -3.0e38f;
  #pragma unroll
  for (int r = 0; r < 8; ++r) {
    const int qr = wq * 16 + lh * 8 + r;      // query row within block (0..63)
    float mx = NEG;
    #pragma unroll
    for (int t = 0; t < 12; ++t) {
      const int j = t * 16 + cl;              // window-relative key col
      const bool wv = (t < 4) ? leftOK : ((t >= 8) ? rightOK : true);
      const bool ok = wv && ((BS + qr) >= j); // causal in window coords
      float v = ok ? s[t][r] * scale : NEG;
      s[t][r] = v;
      mx = fmaxf(mx, v);
    }
    #pragma unroll
    for (int off = 1; off < 16; off <<= 1)
      mx = fmaxf(mx, __shfl_xor(mx, off, 32));
    float sum = 0.f;
    #pragma unroll
    for (int t = 0; t < 12; ++t) {
      float p = __expf(s[t][r] - mx);
      s[t][r] = p;
      sum += p;
    }
    #pragma unroll
    for (int off = 1; off < 16; off <<= 1)
      sum += __shfl_xor(sum, off, 32);
    const float inv = 1.f / sum;
    #pragma unroll
    for (int t = 0; t < 12; ++t)
      P[wq][lh * 8 + r][t * 16 + cl] = (bf16_t)(s[t][r] * inv);
  }
  __syncthreads();   // uniform; orders LDS P writes vs A-layout reads

  // ---- O = P @ vw : A from LDS (row-major P), B from per-block V^T
  v8f o[4] = {};
  const bf16_t* prow = &P[wq][cl][0];
  const bf16_t* vth = Vt + ((size_t)(b * H_ + hh) * NB) * (size_t)(D_ * BS);
  for (int wb = 0; wb < 3; ++wb) {            // window block, uniform guards
    if (wb == 0 && !leftOK)  continue;
    if (wb == 2 && !rightOK) continue;
    const bf16_t* vblk = vth + (size_t)(blk - 1 + wb) * (D_ * BS);
    for (int ks = 0; ks < 2; ++ks) {
      const int jb = wb * BS + ks * 32;
      union { v16bf v; v8bf hl[2]; } a;
      a.hl[0] = *(const v8bf*)(prow + jb + lh * 16);
      a.hl[1] = *(const v8bf*)(prow + jb + lh * 16 + 8);
      #pragma unroll
      for (int n = 0; n < 4; ++n) {
        const bf16_t* vrow = vblk + (size_t)(n * 16 + cl) * BS + ks * 32 + lh * 16;
        union { v16bf v; v8bf hl[2]; } bm;
        bm.hl[0] = *(const v8bf*)(vrow);
        bm.hl[1] = *(const v8bf*)(vrow + 8);
        o[n] = WMMA_BF16(a.v, bm.v, o[n]);
      }
    }
  }

  // ---- store O as bf16 rows [M, C] for the final projection GEMM
  #pragma unroll
  for (int n = 0; n < 4; ++n)
  #pragma unroll
  for (int r = 0; r < 8; ++r) {
    const int tokg = b * N_ + blk * BS + wq * 16 + lh * 8 + r;
    const int ch   = hh * D_ + n * 16 + cl;
    O[(size_t)tokg * C_ + ch] = (bf16_t)o[n][r];
  }
}

// ---------------------------------------------------------------- launch
extern "C" void kernel_launch(void* const* d_in, const int* in_sizes, int n_in,
                              void* d_out, int out_size, void* d_ws, size_t ws_size,
                              hipStream_t stream) {
  const float* x  = (const float*)d_in[0];
  const float* Wq = (const float*)d_in[1];
  const float* bq = (const float*)d_in[2];
  const float* Wk = (const float*)d_in[3];
  const float* bk = (const float*)d_in[4];
  const float* Wv = (const float*)d_in[5];
  const float* bv = (const float*)d_in[6];
  const float* Wo = (const float*)d_in[7];
  const float* bo = (const float*)d_in[8];
  float* out = (float*)d_out;

  // workspace carve-up (all bf16): ~168 MB total
  char* p = (char*)d_ws;
  bf16_t* xb  = (bf16_t*)p; p += (size_t)M_ * K_ * 2;   // 32 MB
  bf16_t* wqb = (bf16_t*)p; p += (size_t)C_ * K_ * 2;   // 2 MB
  bf16_t* wkb = (bf16_t*)p; p += (size_t)C_ * K_ * 2;
  bf16_t* wvb = (bf16_t*)p; p += (size_t)C_ * K_ * 2;
  bf16_t* wob = (bf16_t*)p; p += (size_t)C_ * K_ * 2;
  bf16_t* qb  = (bf16_t*)p; p += (size_t)M_ * K_ * 2;   // [B,H,N,d]
  bf16_t* kb  = (bf16_t*)p; p += (size_t)M_ * K_ * 2;   // [B,H,N,d]
  bf16_t* vt  = (bf16_t*)p; p += (size_t)M_ * K_ * 2;   // [B,H,nb,d,bs]
  bf16_t* ob  = (bf16_t*)p; p += (size_t)M_ * K_ * 2;   // [M,C] attn out

  const int nx = M_ * K_;       // 16,777,216
  const int nw = C_ * K_;       // 1,048,576
  cvt_f32_bf16<<<(nx + 255) / 256, 256, 0, stream>>>(x,  xb,  nx);
  cvt_f32_bf16<<<(nw + 255) / 256, 256, 0, stream>>>(Wq, wqb, nw);
  cvt_f32_bf16<<<(nw + 255) / 256, 256, 0, stream>>>(Wk, wkb, nw);
  cvt_f32_bf16<<<(nw + 255) / 256, 256, 0, stream>>>(Wv, wvb, nw);
  cvt_f32_bf16<<<(nw + 255) / 256, 256, 0, stream>>>(Wo, wob, nw);

  dim3 gg(M_ / 128, C_ / 128);  // 128 x 8 workgroups, 256 thr each
  gemm_wmma<<<gg, 256, 0, stream>>>(xb, wqb, bq, (void*)qb, 0);
  gemm_wmma<<<gg, 256, 0, stream>>>(xb, wkb, bk, (void*)kb, 0);
  gemm_wmma<<<gg, 256, 0, stream>>>(xb, wvb, bv, (void*)vt, 1);

  attn_wmma<<<Bsz * H_ * NB, 128, 0, stream>>>(qb, kb, vt, ob);

  gemm_wmma<<<gg, 256, 0, stream>>>(ob, wob, bo, (void*)out, 2);
}
